// Rnn_net_73134703116636
// MI455X (gfx1250) — compile-verified, matrix operand order checked by reading, not run on hardware
//
#include <hip/hip_runtime.h>
#include <math.h>

typedef _Float16 half_t;
typedef __attribute__((ext_vector_type(16))) _Float16 v16h;
typedef __attribute__((ext_vector_type(8)))  _Float16 v8h;
typedef __attribute__((ext_vector_type(8)))  float    v8f;
typedef __attribute__((ext_vector_type(4)))  unsigned int u32v4;
typedef __attribute__((ext_vector_type(8)))  int i32v8;
typedef __attribute__((ext_vector_type(4)))  int i32v4;

namespace {
constexpr int LAYERS = 10;
constexpr int HID    = 100;
constexpr int HP     = 128;            // hidden padded to 8x16
constexpr int NB     = 512;            // batch
constexpr int NTS    = 512;            // timesteps
constexpr int BT     = 64;             // batch rows per workgroup
constexpr int NCHUNK = NB / BT;        // 8
constexpr int NTILES = HP / 16;        // 8 N-tiles
constexpr int KTILES = HP / 32;        // 4 K-tiles (f16 wmma K=32)

constexpr int FRAG_H = 32 * 16;                   // halfs per packed B fragment
constexpr int MAT_H  = KTILES * NTILES * FRAG_H;  // 16384 halfs / matrix (32 KB)
constexpr int LAY_H  = 2 * MAT_H;                 // per-layer (ih + hh)

// ---- workspace byte offsets ----
constexpr size_t WS_W     = 0;
constexpr size_t WS_W_SZ  = (size_t)LAYERS * LAY_H * sizeof(half_t);   // 640 KB
constexpr size_t WS_BSUM  = WS_W + WS_W_SZ;
constexpr size_t WS_WIH0  = WS_BSUM + (size_t)LAYERS * HP * 4;
constexpr size_t WS_WOUT  = WS_WIH0 + (size_t)HP * 4;
constexpr size_t WS_BOUT  = WS_WOUT + (size_t)HP * 4;
constexpr int    NFLAGS   = 2 * LAYERS * NCHUNK;
constexpr size_t WS_FLAGS = (WS_BOUT + 4 + 255) & ~(size_t)255;
constexpr size_t WS_ACT   = (WS_FLAGS + (size_t)NFLAGS * 4 + 255) & ~(size_t)255;
constexpr size_t ACT_SLOT = (size_t)NB * HP;   // halfs per (layer,parity) slot

// ---- LDS byte offsets inside the single shared block (block base == LDS 0) ----
constexpr unsigned LDS_W    = 0;                                   // 64 KB weights
constexpr unsigned LDS_H    = (unsigned)(LAY_H * sizeof(half_t));  // 16 KB h-state
constexpr unsigned LDS_BSUM = LDS_H + BT * HP * sizeof(half_t);
constexpr unsigned LDS_WIH0 = LDS_BSUM + HP * 4;
constexpr unsigned LDS_WOUT = LDS_WIH0 + HP * 4;
constexpr unsigned LDS_BOUT = LDS_WOUT + HP * 4;
constexpr unsigned LDS_RED  = LDS_BOUT + 16;     // 256 f32 reduction scratch
constexpr unsigned LDS_SZ   = LDS_RED + 256 * 4;
} // namespace

// ---------- fast tanh: native V_TANH_F32 (TRANS32 on CDNA5), branchless fallback ----------
__device__ __forceinline__ float fast_tanh(float x) {
#if __has_builtin(__builtin_amdgcn_tanhf)
  return __builtin_amdgcn_tanhf(x);
#else
  float e = __builtin_amdgcn_exp2f(x * 2.885390081777927f);
  return 1.0f - 2.0f * __builtin_amdgcn_rcpf(e + 1.0f);
#endif
}

// ---------- Tensor Data Mover: linear 1-D transfer descriptor (data_size = 8B) ----------
__device__ __forceinline__ void tdm_desc(u32v4& g0, i32v8& g1, i32v4& g2, i32v4& g3,
                                         unsigned lds_off, const void* gptr, unsigned bytes) {
  unsigned long long ga = (unsigned long long)gptr;
  unsigned n8 = bytes >> 3;                           // 8-byte elements (<= 8192 here)
  g0 = (u32v4)0u;
  g0.x = 1u;                                          // count = 1 valid descriptor
  g0.y = lds_off;                                     // lds_addr (bytes)
  g0.z = (unsigned)ga;                                // global_addr[31:0]
  g0.w = (unsigned)((ga >> 32) & 0x01FFFFFFull) | (2u << 30);   // addr[56:32] | type=2
  g1 = (i32v8)0;
  g1[0] = (int)(3u << 16);                            // data_size = 8 bytes
  g1[1] = (int)((n8 & 0xFFFFu) << 16);                // tensor_dim0[15:0]
  g1[2] = (int)(((n8 >> 16) & 0xFFFFu) | (1u << 16)); // tensor_dim0[31:16] | tensor_dim1=1
  g1[3] = (int)((n8 & 0xFFFFu) << 16);                // tile_dim0 = n8
  g1[4] = 1;                                          // tile_dim1 = 1
  g1[5] = (int)n8;                                    // tensor_dim0_stride[31:0]
  g1[6] = (int)((n8 & 0xFFFFu) << 16);                // | tensor_dim1_stride[15:0]
  g1[7] = (int)(n8 >> 16);                            // tensor_dim1_stride[47:16]
  g2 = (i32v4)0; g2.x = 1;                            // tensor_dim2 = 1
  g3 = (i32v4)0;
}

__device__ __forceinline__ void tdm_load_lds(unsigned lds_off, const void* gptr, unsigned bytes) {
  u32v4 g0; i32v8 g1; i32v4 g2, g3;
  tdm_desc(g0, g1, g2, g3, lds_off, gptr, bytes);
  i32v8 g4 = (i32v8)0;
  __builtin_amdgcn_tensor_load_to_lds(g0, g1, g2, g3, g4, 0);
}
__device__ __forceinline__ void tdm_store_lds(unsigned lds_off, void* gptr, unsigned bytes) {
  u32v4 g0; i32v8 g1; i32v4 g2, g3;
  tdm_desc(g0, g1, g2, g3, lds_off, gptr, bytes);
  i32v8 g4 = (i32v8)0;
  __builtin_amdgcn_tensor_store_from_lds(g0, g1, g2, g3, g4, 0);
}

// ---------- fragment loaders ----------
// A fragment (f16 16x32): lane -> M row; per-lane K runs: hi*8+[0..7] and hi*8+16+[0..7]
__device__ __forceinline__ v16h load_a_frag(const half_t* __restrict__ rowptr, int kt, int hi) {
  const v8h* p0 = (const v8h*)(rowptr + kt * 32 + hi * 8);
  const v8h* p1 = (const v8h*)(rowptr + kt * 32 + hi * 8 + 16);
  v8h lo = *p0, hv = *p1;
  return __builtin_shufflevector(lo, hv, 0,1,2,3,4,5,6,7,8,9,10,11,12,13,14,15);
}
// B fragment: pre-packed 32 lanes x 16 halfs contiguous per (ktile,ntile)
__device__ __forceinline__ v16h load_b_frag(const half_t* __restrict__ wlds, int mat,
                                            int kt, int nt, int lane) {
  const half_t* f = wlds + (size_t)mat * MAT_H + (((kt * NTILES + nt) * 32 + lane) * 16);
  const v8h* p = (const v8h*)f;
  return __builtin_shufflevector(p[0], p[1], 0,1,2,3,4,5,6,7,8,9,10,11,12,13,14,15);
}

// ---------- prep: pack W_ih / W_hh into WMMA B-fragment order (transposed, padded, f16) ----------
__global__ __launch_bounds__(256) void prep_pack(const float* __restrict__ W_ih,
                                                 const float* __restrict__ W_hh,
                                                 half_t* __restrict__ wpack) {
  int id  = blockIdx.x * 256 + threadIdx.x;         // 0 .. 20*16384-1
  int fid = id & (MAT_H - 1);
  int m2  = id >> 14;                               // MAT_H == 2^14
  int l = m2 >> 1, m = m2 & 1;
  int e = fid & 15, lane = (fid >> 4) & 31, nt = (fid >> 9) & 7, kt = fid >> 12;
  int hi = lane >> 4, ln = lane & 15;
  int K = kt * 32 + hi * 8 + (e & 7) + ((e >= 8) ? 16 : 0);
  int N = nt * 16 + ln;
  float v = 0.f;
  if (K < HID && N < HID) {
    if (m == 0) { if (l > 0) v = W_ih[(((l - 1) * HID) + N) * HID + K]; }  // B[k][n]=W[n][k]
    else        { v = W_hh[((l * HID) + N) * HID + K]; }
  }
  wpack[id] = (half_t)v;
}

__global__ __launch_bounds__(256) void prep_misc(const float* __restrict__ W_ih0,
                                                 const float* __restrict__ b_ih,
                                                 const float* __restrict__ b_hh,
                                                 const float* __restrict__ W_out,
                                                 const float* __restrict__ b_out,
                                                 float* __restrict__ bsum,
                                                 float* __restrict__ wih0,
                                                 float* __restrict__ wout,
                                                 float* __restrict__ bout,
                                                 int* __restrict__ flags) {
  int tid = threadIdx.x;
  for (int i = tid; i < LAYERS * HP; i += 256) {
    int l = i / HP, j = i % HP;
    bsum[i] = (j < HID) ? (b_ih[l * HID + j] + b_hh[l * HID + j]) : 0.f;
  }
  for (int j = tid; j < HP; j += 256) {
    wih0[j] = (j < HID) ? W_ih0[j] : 0.f;
    wout[j] = (j < HID) ? W_out[j] : 0.f;
  }
  if (tid == 0) bout[0] = b_out[0];
  for (int i = tid; i < NFLAGS; i += 256) flags[i] = 0;
}

// ---------- persistent layer-pipelined RNN ----------
__global__ __launch_bounds__(256) void rnn_pipe(const float* __restrict__ x,
                                                const float* __restrict__ h0,
                                                float* __restrict__ out,
                                                const half_t* __restrict__ wpack,
                                                const float* __restrict__ bsum,
                                                const float* __restrict__ wih0,
                                                const float* __restrict__ wout,
                                                const float* __restrict__ bout,
                                                int* __restrict__ prod,
                                                int* __restrict__ cons,
                                                half_t* __restrict__ act) {
  __shared__ __align__(128) char smem[LDS_SZ];
  half_t* wlds   = (half_t*)(smem + LDS_W);
  half_t* hlds   = (half_t*)(smem + LDS_H);
  float*  bsum_l = (float*)(smem + LDS_BSUM);
  float*  wih0_l = (float*)(smem + LDS_WIH0);
  float*  wout_l = (float*)(smem + LDS_WOUT);
  float*  bout_l = (float*)(smem + LDS_BOUT);
  float*  red_l  = (float*)(smem + LDS_RED);

  const int l   = blockIdx.x / NCHUNK;
  const int ch  = blockIdx.x % NCHUNK;
  const int b0  = ch * BT;
  const int tid = threadIdx.x;
  const int wv = tid >> 5, lane = tid & 31;
  const int mi = wv >> 1, ngbase = (wv & 1) * 4;
  const int hi = lane >> 4, ln = lane & 15;

  // one-time: DMA this layer's packed weights (64 KB) into LDS via the TDM
  if (wv == 0) {
    tdm_load_lds(LDS_W, wpack + (size_t)l * LAY_H, (unsigned)(LAY_H * sizeof(half_t)));
    __builtin_amdgcn_s_wait_tensorcnt(0);
  }
  for (int i = tid; i < HP; i += 256) {
    bsum_l[i] = bsum[l * HP + i];
    wih0_l[i] = (l == 0) ? wih0[i] : 0.f;
    wout_l[i] = (l == LAYERS - 1) ? wout[i] : 0.f;
  }
  if (tid == 0) bout_l[0] = bout[0];
  for (int i = tid; i < BT * HP; i += 256) {   // init hidden state (f16, padded)
    int r = i / HP, c = i % HP;
    float v = (c < HID) ? h0[((size_t)l * NB + (b0 + r)) * HID + c] : 0.f;
    hlds[i] = (half_t)v;
  }
  __syncthreads();

  // hoist all loop-invariant B fragments into registers:
  // 2 matrices x 4 N-tiles x 4 K-tiles = 32 fragments (256 VGPRs)
  v16h bih[4][KTILES], bhh[4][KTILES];
  float bj_r[4], w0_r[4];
  #pragma unroll
  for (int g = 0; g < 4; ++g) {
    #pragma unroll
    for (int kt = 0; kt < KTILES; ++kt) {
      bih[g][kt] = load_b_frag(wlds, 0, kt, ngbase + g, lane);
      bhh[g][kt] = load_b_frag(wlds, 1, kt, ngbase + g, lane);
    }
    const int j = (ngbase + g) * 16 + ln;
    bj_r[g] = bsum_l[j];          // loop-invariant bias (col-indexed)
    w0_r[g] = wih0_l[j];          // loop-invariant layer-0 input weight
  }

  for (int t = 0; t < NTS; ++t) {
    const int p = t & 1;

    v16h ax[KTILES];
    if (l > 0) {                                   // wait for producer layer l-1, time t
      const int* pf = prod + (l - 1) * NCHUNK + ch;
      while (__hip_atomic_load(pf, __ATOMIC_RELAXED, __HIP_MEMORY_SCOPE_AGENT) < t + 1)
        __builtin_amdgcn_s_sleep(1);
      __builtin_amdgcn_fence(__ATOMIC_ACQUIRE, "agent");
      const half_t* arow = act + ((size_t)((l - 1) * 2 + p)) * ACT_SLOT
                               + (size_t)(b0 + mi * 16 + ln) * HP;
      #pragma unroll
      for (int kt = 0; kt < KTILES; ++kt) ax[kt] = load_a_frag(arow, kt, hi);
    }
    v16h ah[KTILES];
    const half_t* hrow = hlds + (mi * 16 + ln) * HP;
    #pragma unroll
    for (int kt = 0; kt < KTILES; ++kt) ah[kt] = load_a_frag(hrow, kt, hi);

    float xr[8];
    if (l == 0) {
      #pragma unroll
      for (int r = 0; r < 8; ++r)
        xr[r] = x[(size_t)(b0 + mi * 16 + hi * 8 + r) * NTS + t];
    }

    float hnew[4][8];
    #pragma unroll
    for (int g = 0; g < 4; ++g) {
      const float bj = bj_r[g];
      v8f c;
      if (l == 0) {
        const float w0 = w0_r[g];
        #pragma unroll
        for (int r = 0; r < 8; ++r) c[r] = bj + xr[r] * w0;   // K=1 input path folded in
      } else {
        #pragma unroll
        for (int r = 0; r < 8; ++r) c[r] = bj;
      }
      if (l > 0) {
        #pragma unroll
        for (int kt = 0; kt < KTILES; ++kt)
          c = __builtin_amdgcn_wmma_f32_16x16x32_f16(false, ax[kt], false, bih[g][kt],
                                                     (short)0, c, false, false);
      }
      #pragma unroll
      for (int kt = 0; kt < KTILES; ++kt)
        c = __builtin_amdgcn_wmma_f32_16x16x32_f16(false, ah[kt], false, bhh[g][kt],
                                                   (short)0, c, false, false);
      #pragma unroll
      for (int r = 0; r < 8; ++r) hnew[g][r] = fast_tanh(c[r]);
    }

    __syncthreads();                               // all reads of old h done
    if (tid == 0 && l > 0)
      __hip_atomic_store(cons + (l - 1) * NCHUNK + ch, t + 1,
                         __ATOMIC_RELEASE, __HIP_MEMORY_SCOPE_AGENT);

    #pragma unroll
    for (int g = 0; g < 4; ++g) {
      const int nt = ngbase + g;
      #pragma unroll
      for (int r = 0; r < 8; ++r)
        hlds[(mi * 16 + hi * 8 + r) * HP + nt * 16 + ln] = (half_t)hnew[g][r];
    }
    __syncthreads();                               // new h visible to whole WG

    if (l < LAYERS - 1) {
      if (wv == 0) {                               // publish h(t): TDM store 16 KB to L2 slot
        if (t >= 2) {
          const int* cf = cons + l * NCHUNK + ch;
          while (__hip_atomic_load(cf, __ATOMIC_RELAXED, __HIP_MEMORY_SCOPE_AGENT) < t - 1)
            __builtin_amdgcn_s_sleep(1);
          __builtin_amdgcn_fence(__ATOMIC_ACQUIRE, "agent");
        }
        tdm_store_lds(LDS_H, act + ((size_t)(l * 2 + p)) * ACT_SLOT + (size_t)b0 * HP,
                      (unsigned)(BT * HP * sizeof(half_t)));
        __builtin_amdgcn_s_wait_tensorcnt(0);
        if (lane == 0)
          __hip_atomic_store(prod + l * NCHUNK + ch, t + 1,
                             __ATOMIC_RELEASE, __HIP_MEMORY_SCOPE_AGENT);
      }
    } else {
      // final Linear y = h9 . W_out + b_out, parallel over all 256 threads:
      // row = tid>>2 (64 rows), quarter q = tid&3 covers 25 of the 100 columns
      const int row = tid >> 2, q = tid & 3;
      const half_t* hr = hlds + row * HP + q * 25;
      const float*  wq = wout_l + q * 25;
      float pp = 0.f;
      #pragma unroll 5
      for (int j = 0; j < 25; ++j) pp += (float)hr[j] * wq[j];
      red_l[tid] = pp;
      __syncthreads();
      if (tid < BT) {
        float acc = bout_l[0] + red_l[tid * 4] + red_l[tid * 4 + 1]
                              + red_l[tid * 4 + 2] + red_l[tid * 4 + 3];
        out[(size_t)(b0 + tid) * NTS + t] = acc;
      }
    }
  }

  __syncthreads();                                 // write h_final (fp32) after outs region
  for (int i = tid; i < BT * HID; i += 256) {
    int r = i / HID, c = i % HID;
    out[(size_t)NB * NTS + ((size_t)l * NB + (b0 + r)) * HID + c] = (float)hlds[r * HP + c];
  }
}

extern "C" void kernel_launch(void* const* d_in, const int* in_sizes, int n_in,
                              void* d_out, int out_size, void* d_ws, size_t ws_size,
                              hipStream_t stream) {
  const float* x      = (const float*)d_in[0];
  const float* h0     = (const float*)d_in[1];
  const float* W_ih0  = (const float*)d_in[2];
  const float* W_ih   = (const float*)d_in[3];
  const float* W_hh   = (const float*)d_in[4];
  const float* b_ih   = (const float*)d_in[5];
  const float* b_hh   = (const float*)d_in[6];
  const float* W_out  = (const float*)d_in[7];
  const float* b_out  = (const float*)d_in[8];
  float* out = (float*)d_out;
  char*  ws  = (char*)d_ws;

  half_t* wpack = (half_t*)(ws + WS_W);
  float*  bsum  = (float*)(ws + WS_BSUM);
  float*  wih0  = (float*)(ws + WS_WIH0);
  float*  wout  = (float*)(ws + WS_WOUT);
  float*  bout  = (float*)(ws + WS_BOUT);
  int*    flags = (int*)(ws + WS_FLAGS);
  int*    prod  = flags;
  int*    cons  = flags + LAYERS * NCHUNK;
  half_t* act   = (half_t*)(ws + WS_ACT);

  (void)in_sizes; (void)n_in; (void)out_size; (void)ws_size;

  prep_pack<<<(LAYERS * LAY_H) / 256, 256, 0, stream>>>(W_ih, W_hh, wpack);
  prep_misc<<<1, 256, 0, stream>>>(W_ih0, b_ih, b_hh, W_out, b_out,
                                   bsum, wih0, wout, bout, flags);
  rnn_pipe<<<LAYERS * NCHUNK, 256, 0, stream>>>(x, h0, out, wpack, bsum, wih0, wout,
                                                bout, prod, cons, act);
}
